// GATLayer_76166950028260
// MI455X (gfx1250) — matrix operand order repaired and checked by reading, not compiled, not run on hardware
//
#include <hip/hip_runtime.h>

typedef __attribute__((ext_vector_type(16))) __bf16 v16bf;
typedef __attribute__((ext_vector_type(8)))  float  v8f;
typedef __attribute__((ext_vector_type(4)))  unsigned int v4u;
typedef __attribute__((ext_vector_type(8)))  int v8i;
typedef __attribute__((ext_vector_type(4)))  int v4i;

#define B_TOT 16384
#define NNB   20
#define FDIM  128
#define TDIM  100
#define EDIM  128
#define KEYD  356
#define KEYDP 384
#define BT    16

// raw fp32 staging layout (per double-buffer): [nbr 16x128][edge 16x128][time 16x100]
#define RAWN  0
#define RAWE  (BT * FDIM)
#define RAWT  (2 * BT * FDIM)
#define RAWSZ (2 * BT * FDIM + BT * TDIM)   // 5696 floats = 22784 B

// ---- weight prep: transpose/convert to bf16, pre-swizzled into WMMA B-fragment order.
// B-frag (16x16x32 bf16): lane<16 -> N=lane, K=kt*32+(j<8 ? j : 16+j-8) ; lane>=16 -> N=lane-16, K=that+8.
// Stored so each lane's 16 halves are contiguous: D[((kt*ntiles+nt)*32+lane)*16+j].
__global__ void prep_weights_kernel(const float* __restrict__ S, __bf16* __restrict__ D,
                                    int K, int N, int Kreal, int srcLD, int transpose) {
  int total = K * N;
  int ntiles = N >> 4;
  for (int i = blockIdx.x * blockDim.x + threadIdx.x; i < total; i += gridDim.x * blockDim.x) {
    int j    = i & 15;
    int lane = (i >> 4) & 31;
    int t    = i >> 9;
    int nt   = t % ntiles;
    int kt   = t / ntiles;
    int k = kt * 32 + ((lane >> 4) << 3) + (j & 7) + ((j >> 3) << 4);
    int n = nt * 16 + (lane & 15);
    float v = 0.f;
    if (k < Kreal) v = transpose ? S[n * srcLD + k] : S[k * srcLD + n];
    D[i] = (__bf16)v;
  }
}

// A-frag (16x32 bf16): lane<16: row=lane, K j=0..7 -> j, j=8..15 -> 16+(j-8); lane>=16: +8.
__device__ __forceinline__ v16bf load_a_frag(const __bf16* base, int ldm, int kbase) {
  int lane = threadIdx.x & 31;
  const __bf16* p = base + (lane & 15) * ldm + kbase + ((lane & 16) >> 1);
  v16bf a;
  ((uint4*)&a)[0] = *(const uint4*)p;
  ((uint4*)&a)[1] = *(const uint4*)(p + 16);
  return a;
}

__device__ __forceinline__ v16bf load_b_frag(const __bf16* mat, int ntiles, int kt, int nt) {
  int lane = threadIdx.x & 31;
  const __bf16* p = mat + (((kt * ntiles + nt) * 32 + lane) << 4);
  v16bf b;
  ((uint4*)&b)[0] = ((const uint4*)p)[0];
  ((uint4*)&b)[1] = ((const uint4*)p)[1];
  return b;
}

template <int NCH>
__device__ __forceinline__ v8f gemm_tile(const __bf16* Alds, int lda,
                                         const __bf16* Bmat, int ntilesB,
                                         int kt0, int nt) {
  v8f acc = {};
#pragma unroll
  for (int kc = 0; kc < NCH; ++kc) {
    v16bf a = load_a_frag(Alds, lda, kc * 32);
    v16bf b = load_b_frag(Bmat, ntilesB, kt0 + kc, nt);
    acc = __builtin_amdgcn_wmma_f32_16x16x32_bf16(false, a, false, b, (short)0, acc,
                                                  false, false);
  }
  return acc;
}

// ---- TDM: 2D tile (16 rows x rowlen f32, row stride rowstride elems) -> LDS.
// D# per CDNA5 ISA 8.3/8.4: group0 = {count=1, lds_addr, global_addr | type=2};
// group1 = {data_size=4B, tensor_dim0/1, tile_dim0/1, tensor_dim0_stride}.
__device__ __forceinline__ void tdm_load_2d(unsigned lds_off, const float* gbase,
                                            long long start_elem, int rowlen,
                                            int rowstride) {
  unsigned long long ga =
      (unsigned long long)(uintptr_t)gbase + (unsigned long long)start_elem * 4ull;
  v4u g0;
  g0[0] = 1u;                                            // count=1, no gather
  g0[1] = lds_off;                                       // LDS byte address
  g0[2] = (unsigned)(ga & 0xffffffffu);                  // global_addr[31:0]
  g0[3] = (unsigned)((ga >> 32) & 0x1ffffffu) | (2u << 30);  // addr[56:32] | type=2
  v8i g1;
  g1[0] = (int)(2u << 16);                               // data_size = 4 bytes
  g1[1] = (int)((unsigned)(rowlen & 0xffff) << 16);      // tensor_dim0[15:0]
  g1[2] = (int)(((unsigned)rowlen >> 16) | (16u << 16)); // tdim0 hi | tensor_dim1=16
  g1[3] = (int)((unsigned)(rowlen & 0xffff) << 16);      // tile_dim0 = rowlen
  g1[4] = 16;                                            // tile_dim1 = 16, tile_dim2 = 0
  g1[5] = rowstride;                                     // tensor_dim0_stride[31:0]
  g1[6] = 0;
  g1[7] = 0;
  v4i z4 = {0, 0, 0, 0};                                 // 2D: groups 2/3 zero
  v8i z8 = {0, 0, 0, 0, 0, 0, 0, 0};
  __builtin_amdgcn_tensor_load_to_lds(g0, g1, z4, z4, z8, 0);
}

__global__ __launch_bounds__(256, 1)
void gat_fused_kernel(const float* __restrict__ src_g,
                      const float* __restrict__ nbr_g,
                      const float* __restrict__ time_g,
                      const float* __restrict__ edge_g,
                      const unsigned char* __restrict__ mask_g,
                      const float* __restrict__ wgt_g,
                      const float* __restrict__ bt,  const float* __restrict__ bq,
                      const float* __restrict__ bk,  const float* __restrict__ bv,
                      const float* __restrict__ bo,  const float* __restrict__ b1,
                      const float* __restrict__ b2,
                      const __bf16* __restrict__ WtB, const __bf16* __restrict__ WqB,
                      const __bf16* __restrict__ WkB, const __bf16* __restrict__ WvB,
                      const __bf16* __restrict__ WoB, const __bf16* __restrict__ W1B,
                      const __bf16* __restrict__ W2B,
                      float* __restrict__ out_g) {
  // keyv stored n-major: row index = n*16 + b  (no div/mod anywhere)
  __shared__ __align__(16) __bf16 s_keyv[NNB * BT][FDIM];   // 80 KB
  __shared__ __align__(16) float  s_raw[2][RAWSZ];          // 44.5 KB TDM double buffer
  __shared__ __align__(16) __bf16 s_atile[BT][KEYDP];       // 12 KB
  __shared__ __align__(16) __bf16 s_src[BT][FDIM];
  __shared__ __align__(16) __bf16 s_q[BT][FDIM];
  __shared__ __align__(16) __bf16 s_qk[2][BT][FDIM];
  __shared__ __align__(16) __bf16 s_sv[2][BT][FDIM];
  __shared__ __align__(16) __bf16 s_ctx[BT][FDIM];
  __shared__ __align__(16) __bf16 s_cat[BT][2 * FDIM];
  __shared__ __align__(16) __bf16 s_hmid[BT][FDIM];
  __shared__ float s_sc[BT][2][NNB];
  __shared__ float s_wgt[BT][NNB];
  __shared__ unsigned char s_msk[BT][NNB];
  __shared__ int s_inv[BT];

  const int tid   = threadIdx.x;
  const int wave  = tid >> 5;
  const int lane  = tid & 31;
  const int bb    = blockIdx.x;
  const int col16 = lane & 15;
  const int mbase = (lane & 16) >> 1;  // C-frag: lanes 0-15 -> M=r, lanes 16-31 -> M=8+r

  // ---- stage 0: stage src / weights / mask; wave0 kicks off TDM for chunk 0
  if (wave == 0 && lane == 0) {
    long long row0 = (long long)bb * (BT * NNB);   // first (b,n) row of this block
    tdm_load_2d((unsigned)(uintptr_t)&s_raw[0][RAWN], nbr_g,  (row0 + 0) * FDIM, FDIM, NNB * FDIM);
    tdm_load_2d((unsigned)(uintptr_t)&s_raw[0][RAWE], edge_g, (row0 + 0) * EDIM, EDIM, NNB * EDIM);
    tdm_load_2d((unsigned)(uintptr_t)&s_raw[0][RAWT], time_g, (row0 + 0) * TDIM, TDIM, NNB * TDIM);
  }
  for (int i = tid; i < BT * FDIM; i += 256) {
    int b = i >> 7, f = i & 127;
    __bf16 h = (__bf16)src_g[(bb * BT + b) * FDIM + f];
    s_src[b][f] = h;
    s_cat[b][FDIM + f] = h;  // second half of merge input
  }
  for (int i = tid; i < BT * NNB; i += 256) {
    int b = i / NNB, n = i % NNB;
    s_wgt[b][n] = wgt_g[(bb * BT + b) * NNB + n];
    s_msk[b][n] = mask_g[(bb * BT + b) * NNB + n];
  }
  __syncthreads();

  // ---- stage 1: keyv = (concat(nbr,edge,time) @ Wt^T + bt) * w  (bf16 in LDS)
  // chunk c = neighbor index c for all 16 batch rows; TDM double-buffered.
  for (int c = 0; c < NNB; ++c) {
    const int cur = c & 1, nxt = cur ^ 1;
    if (wave == 0) {
      if (c + 1 < NNB) {
        if (lane == 0) {
          long long row0 = (long long)bb * (BT * NNB) + (c + 1);
          tdm_load_2d((unsigned)(uintptr_t)&s_raw[nxt][RAWN], nbr_g,  row0 * FDIM, FDIM, NNB * FDIM);
          tdm_load_2d((unsigned)(uintptr_t)&s_raw[nxt][RAWE], edge_g, row0 * EDIM, EDIM, NNB * EDIM);
          tdm_load_2d((unsigned)(uintptr_t)&s_raw[nxt][RAWT], time_g, row0 * TDIM, TDIM, NNB * TDIM);
        }
        __builtin_amdgcn_s_wait_tensorcnt(3);  // chunk c done; c+1 still in flight
      } else {
        __builtin_amdgcn_s_wait_tensorcnt(0);
      }
    }
    __syncthreads();
    // convert fp32 raw -> bf16 A-tile (2 rows per wave, 12 cols per lane; no div/mod)
    {
      const float* raw = s_raw[cur];
#pragma unroll
      for (int rr = 0; rr < 2; ++rr) {
        int r = wave * 2 + rr;
#pragma unroll
        for (int cc = 0; cc < 12; ++cc) {
          int col = cc * 32 + lane;
          float v;
          if (col < 128)       v = raw[RAWN + r * FDIM + col];
          else if (col < 256)  v = raw[RAWE + r * EDIM + (col - 128)];
          else if (col < KEYD) v = raw[RAWT + r * TDIM + (col - 256)];
          else                 v = 0.f;
          s_atile[r][col] = (__bf16)v;
        }
      }
    }
    __syncthreads();
    {
      v8f acc = gemm_tile<12>(&s_atile[0][0], KEYDP, WtB, 8, 0, wave);
      int colg = wave * 16 + col16;
      float bias = bt[colg];
#pragma unroll
      for (int r = 0; r < 8; ++r) {
        int b = mbase + r;
        s_keyv[c * BT + b][colg] = (__bf16)((acc[r] + bias) * s_wgt[b][c]);
      }
    }
    __syncthreads();
  }

  // ---- stage 2: q = src @ Wq^T + bq
  {
    v8f acc = gemm_tile<4>(&s_src[0][0], FDIM, WqB, 8, 0, wave);
    int colg = wave * 16 + col16;
    float bias = bq[colg];
#pragma unroll
    for (int r = 0; r < 8; ++r) s_q[mbase + r][colg] = (__bf16)(acc[r] + bias);
  }
  __syncthreads();

  // ---- stage 3: qk[h] = q_h @ Wk[h*64:(h+1)*64, :]
#pragma unroll
  for (int h = 0; h < 2; ++h) {
    v8f acc = gemm_tile<2>(&s_q[0][h * 64], FDIM, WkB, 8, h * 2, wave);
    int colg = wave * 16 + col16;
#pragma unroll
    for (int r = 0; r < 8; ++r) s_qk[h][mbase + r][colg] = (__bf16)acc[r];
  }
  __syncthreads();

  // ---- stage 4: raw scores (dot of qk_h and keyv rows)
  for (int i = tid; i < BT * 2 * NNB; i += 256) {
    int b = i / (2 * NNB);
    int h = (i / NNB) & 1;
    int n = i % NNB;
    const __bf16* qr = s_qk[h][b];
    const __bf16* kr = s_keyv[n * BT + b];
    float d = 0.f;
#pragma unroll 8
    for (int c = 0; c < FDIM; ++c) d += (float)qr[c] * (float)kr[c];
    s_sc[b][h][n] = d;
  }
  __syncthreads();

  // masked softmax with invalid-row fixup; bk enters as constant q_h . bk_h
  if (tid < BT * 2) {
    int b = tid >> 1, h = tid & 1;
    float qb = 0.f;
    for (int j = 0; j < 64; ++j) qb += (float)s_q[b][h * 64 + j] * bk[h * 64 + j];
    bool inv = true;
    for (int n = 0; n < NNB; ++n) inv = inv && (s_msk[b][n] != 0);
    if (h == 0) s_inv[b] = inv ? 1 : 0;
    const float scale = 0.125f;  // 1/sqrt(64)
    float sc[NNB];
    float m = -3.4e38f;
#pragma unroll
    for (int n = 0; n < NNB; ++n) {
      bool pad = (s_msk[b][n] != 0) && !(inv && n == 0);
      float v = pad ? -3.4e38f : (s_sc[b][h][n] + qb) * scale;
      sc[n] = v;
      m = fmaxf(m, v);
    }
    float sum = 0.f;
#pragma unroll
    for (int n = 0; n < NNB; ++n) {
      float e = (sc[n] <= -3.0e38f) ? 0.f : __expf(sc[n] - m);
      sc[n] = e;
      sum += e;
    }
    float rinv = 1.f / sum;
#pragma unroll
    for (int n = 0; n < NNB; ++n) s_sc[b][h][n] = sc[n] * rinv;
  }
  __syncthreads();

  // ---- stage 5: sv[h] = attn_h @ keyv
  for (int i = tid; i < 2 * BT * FDIM; i += 256) {
    int h = i >> 11, b = (i >> 7) & 15, c = i & 127;
    float acc = 0.f;
#pragma unroll
    for (int n = 0; n < NNB; ++n) acc += s_sc[b][h][n] * (float)s_keyv[n * BT + b][c];
    s_sv[h][b][c] = (__bf16)acc;
  }
  __syncthreads();

  // ---- stage 6: ctx[:, h*64+j] = sv[h] @ Wv_h^T + bv  (softmax sums to 1)
  {
    int h = wave >> 2, st = wave & 3;
    v8f acc = gemm_tile<4>(&s_sv[h][0][0], FDIM, WvB, 8, 0, h * 4 + st);
    int colg = h * 64 + st * 16 + col16;
    float bias = bv[colg];
#pragma unroll
    for (int r = 0; r < 8; ++r) s_ctx[mbase + r][colg] = (__bf16)(acc[r] + bias);
  }
  __syncthreads();

  // ---- stage 7: out = relu(ctx @ Wo^T + bo), zero invalid -> cat[:, :128]
  {
    v8f acc = gemm_tile<4>(&s_ctx[0][0], FDIM, WoB, 8, 0, wave);
    int colg = wave * 16 + col16;
    float bias = bo[colg];
#pragma unroll
    for (int r = 0; r < 8; ++r) {
      int M = mbase + r;
      float v = fmaxf(acc[r] + bias, 0.f);
      if (s_inv[M]) v = 0.f;
      s_cat[M][colg] = (__bf16)v;
    }
  }
  __syncthreads();

  // ---- stage 8: hmid = relu(cat @ W1^T + b1)   (K = 256)
  {
    v8f acc = gemm_tile<8>(&s_cat[0][0], 2 * FDIM, W1B, 8, 0, wave);
    int colg = wave * 16 + col16;
    float bias = b1[colg];
#pragma unroll
    for (int r = 0; r < 8; ++r)
      s_hmid[mbase + r][colg] = (__bf16)fmaxf(acc[r] + bias, 0.f);
  }
  __syncthreads();

  // ---- stage 9: result = hmid @ W2^T + b2  -> fp32 global
  {
    v8f acc = gemm_tile<4>(&s_hmid[0][0], FDIM, W2B, 8, 0, wave);
    int colg = wave * 16 + col16;
    float bias = b2[colg];
#pragma unroll
    for (int r = 0; r < 8; ++r)
      out_g[(bb * BT + mbase + r) * FDIM + colg] = acc[r] + bias;
  }
}

extern "C" void kernel_launch(void* const* d_in, const int* in_sizes, int n_in,
                              void* d_out, int out_size, void* d_ws, size_t ws_size,
                              hipStream_t stream) {
  (void)in_sizes; (void)n_in; (void)out_size; (void)ws_size;
  const float* src  = (const float*)d_in[0];
  const float* nbr  = (const float*)d_in[1];
  const float* timf = (const float*)d_in[2];
  const float* edgf = (const float*)d_in[3];
  const unsigned char* msk = (const unsigned char*)d_in[4];
  const float* wgt  = (const float*)d_in[5];
  const float* Wt = (const float*)d_in[6];  const float* bt = (const float*)d_in[7];
  const float* Wq = (const float*)d_in[8];  const float* bq = (const float*)d_in[9];
  const float* Wk = (const float*)d_in[10]; const float* bk = (const float*)d_in[11];
  const float* Wv = (const float*)d_in[12]; const float* bv = (const float*)d_in[13];
  const float* Wo = (const float*)d_in[14]; const float* bo = (const float*)d_in[15];
  const float* W1 = (const float*)d_in[16]; const float* b1 = (const float*)d_in[17];
  const float* W2 = (const float*)d_in[18]; const float* b2 = (const float*)d_in[19];
  float* out = (float*)d_out;

  __bf16* ws  = (__bf16*)d_ws;
  __bf16* WtB = ws;                    // [384 x 128] (K zero-padded 356->384)
  __bf16* WqB = WtB + 384 * 128;
  __bf16* WkB = WqB + 128 * 128;
  __bf16* WvB = WkB + 128 * 128;
  __bf16* WoB = WvB + 128 * 128;
  __bf16* W1B = WoB + 128 * 128;       // [256 x 128]
  __bf16* W2B = W1B + 256 * 128;

  prep_weights_kernel<<<192, 256, 0, stream>>>(Wt, WtB, 384, 128, 356, 356, 1);
  prep_weights_kernel<<<64, 256, 0, stream>>>(Wq, WqB, 128, 128, 128, 128, 1);
  prep_weights_kernel<<<64, 256, 0, stream>>>(Wk, WkB, 128, 128, 128, 128, 0);
  prep_weights_kernel<<<64, 256, 0, stream>>>(Wv, WvB, 128, 128, 128, 128, 1);
  prep_weights_kernel<<<64, 256, 0, stream>>>(Wo, WoB, 128, 128, 128, 128, 1);
  prep_weights_kernel<<<64, 256, 0, stream>>>(W1, W1B, 256, 128, 256, 256, 1);
  prep_weights_kernel<<<64, 256, 0, stream>>>(W2, W2B, 128, 128, 128, 128, 1);

  gat_fused_kernel<<<B_TOT / BT, 256, 0, stream>>>(
      src, nbr, timf, edgf, msk, wgt, bt, bq, bk, bv, bo, b1, b2,
      WtB, WqB, WkB, WvB, WoB, W1B, W2B, out);
}
